// LSTM_AE_CLASSIFIER_V4_Experimental_35510789604108
// MI455X (gfx1250) — matrix operand order repaired and checked by reading, not compiled, not run on hardware
//
#include <hip/hip_runtime.h>
#include <hip/hip_bf16.h>

// ---------------------------------------------------------------------------
// Fused LSTM autoencoder + classifier for MI455X (gfx1250, wave32, WMMA).
// One pass over T: enc_rec, enc_clas, decoder pipelined per timestep.
// v4: single rolled time-step (no ×2 inline), x prefetch via 14 regs +
//     LDS-double-buffered x tile (halves pipeline registers), per-gate
//     fragment loads (no big preload windows), waves_per_eu(4) cap.
//     Goal: fit 256 VGPRs with zero spills and zero MSB prefixes.
// ---------------------------------------------------------------------------

#define BATCH   32768
#define TSTEPS  28
#define IDIM    28
#define HDIM    16
#define NCLS    10
#define KPAD    32
#define TILE_M  16
#define NWAVES  8
#define BLOCK_THREADS (NWAVES * 32)
#define XREG    14                  // (16*28)/32 x-elements per lane

// LDS layout (bytes)
#define OFF_WIH_R 0u        // [64][32] f16 = 4096
#define OFF_WHH_R 4096u
#define OFF_WIH_C 8192u
#define OFF_WHH_C 12288u
#define OFF_WIH_D 16384u    // [112][32] f16 = 7168
#define OFF_WHH_D 23552u
#define OFF_BIAS_R 30720u   // [64] f32
#define OFF_BIAS_C 30976u
#define OFF_BIAS_D 31232u   // [112] f32, padded to 512
#define OFF_STAGE  31744u
// per wave: xa0 1K | xa1 1K | ha_r 1K | ha_c 1K | ha_d 1K | gates 7K | c_d 1.75K
#define STAGE_PER_WAVE 14080u
#define SMEM_BYTES (OFF_STAGE + NWAVES * STAGE_PER_WAVE)   // 144384 (2 blocks/WGP)

typedef __attribute__((ext_vector_type(16))) _Float16 v16h;
typedef __attribute__((ext_vector_type(8)))  _Float16 v8h;
typedef __attribute__((ext_vector_type(8)))  float    v8f;

__device__ __forceinline__ v8f splat8(float v) {
    v8f r;
    #pragma unroll
    for (int k = 0; k < 8; ++k) r[k] = v;
    return r;
}

__device__ __forceinline__ float sigm(float x) {
    return 1.0f / (1.0f + __expf(-x));
}
__device__ __forceinline__ float tanh_fast(float x) {
    // tanh(x) = 2*sigmoid(2x) - 1  (single v_exp_f32)
    return 2.0f / (1.0f + __expf(-2.0f * x)) - 1.0f;
}

__device__ __forceinline__ v8f wmma_f16(v16h a, v16h b, v8f c) {
    // D = A(16x32 f16) * B(32x16 f16) + C(16x16 f32)
    return __builtin_amdgcn_wmma_f32_16x16x32_f16(
        /*neg_a=*/false, a, /*neg_b=*/false, b,
        /*c_mod=*/(short)0, c, /*reuse_a=*/false, /*reuse_b=*/false);
}

// A fragment (16x32, f16) from row-major LDS tile [16][KPAD].
// Lane L<16 holds row L, K={8*hi..8*hi+7} and {16+8*hi..}, hi = L>>4.
__device__ __forceinline__ v16h load_A_frag(const _Float16* base, int lane) {
    int hi  = lane >> 4;
    int row = lane & 15;
    const v8h* p0 = (const v8h*)(base + row * KPAD + hi * 8);
    const v8h* p1 = (const v8h*)(base + row * KPAD + 16 + hi * 8);
    v8h lo = *p0, hh = *p1;
    return __builtin_shufflevector(lo, hh, 0,1,2,3,4,5,6,7,8,9,10,11,12,13,14,15);
}

// B fragment (32x16, f16) from weight LDS stored as [N][KPAD] (K contiguous).
// Lane L: column N=n_base+(L&15), K={8*hi..8*hi+7} and {16+8*hi..}, hi = L>>4.
__device__ __forceinline__ v16h load_B_frag(const _Float16* wp, int n_base, int lane) {
    int n  = n_base + (lane & 15);
    int kb = (lane >> 4) * 8;
    const v8h* p0 = (const v8h*)(wp + n * KPAD + kb);
    const v8h* p1 = (const v8h*)(wp + n * KPAD + kb + 16);
    v8h lo = *p0, hh = *p1;
    return __builtin_shufflevector(lo, hh, 0,1,2,3,4,5,6,7,8,9,10,11,12,13,14,15);
}

// Store a C-fragment (f32) as f16 into an A-staging tile [16][KPAD].
// C layout: VGPR k, lane L -> (M = k + 8*(L>=16), N = L&15).
__device__ __forceinline__ void store_h_f16(_Float16* base, int lane, v8f h) {
    int row0 = (lane >> 4) * 8;
    int n    = lane & 15;
    #pragma unroll
    for (int k = 0; k < 8; ++k)
        base[(row0 + k) * KPAD + n] = (_Float16)h[k];
}

// Fetch this lane's 14 x-elements for timestep t (coalesced along i).
__device__ __forceinline__ void fetch_x(const float* __restrict__ x, int b0, int t,
                                        int lane, float xr[XREG]) {
    #pragma unroll
    for (int j = 0; j < XREG; ++j) {
        int idx = lane + j * 32;
        int r = idx / IDIM, i = idx - r * IDIM;      // t-invariant (LICM'd)
        xr[j] = x[((size_t)(b0 + r) * TSTEPS + t) * IDIM + i];
    }
}

// Stage the register x-tile into LDS as f16 (cols 28..31 remain zero).
__device__ __forceinline__ void stage_x(_Float16* xa, int lane, const float xr[XREG]) {
    #pragma unroll
    for (int j = 0; j < XREG; ++j) {
        int idx = lane + j * 32;
        int r = idx / IDIM, i = idx - r * IDIM;
        xa[r * KPAD + i] = (_Float16)xr[j];
    }
}

// One encoder LSTM step for a 16-row tile (H=16: gate tiles align with gates).
// Per-gate fragment loads keep transient pressure low; the scheduler hoists
// them ahead of the WMMAs when registers allow.
__device__ __forceinline__ void enc_step(v16h aX, _Float16* ha,
                                         const _Float16* wih, const _Float16* whh,
                                         const float bias[4], v8f& c, int lane) {
    v16h aH = load_A_frag(ha, lane);           // h_{t-1} (read before overwrite)
    v8f gi = splat8(bias[0]);
    v8f gf = splat8(bias[1]);
    v8f gg = splat8(bias[2]);
    v8f go = splat8(bias[3]);
    gi = wmma_f16(aX, load_B_frag(wih, 0,  lane), gi);
    gi = wmma_f16(aH, load_B_frag(whh, 0,  lane), gi);
    gf = wmma_f16(aX, load_B_frag(wih, 16, lane), gf);
    gf = wmma_f16(aH, load_B_frag(whh, 16, lane), gf);
    gg = wmma_f16(aX, load_B_frag(wih, 32, lane), gg);
    gg = wmma_f16(aH, load_B_frag(whh, 32, lane), gg);
    go = wmma_f16(aX, load_B_frag(wih, 48, lane), go);
    go = wmma_f16(aH, load_B_frag(whh, 48, lane), go);
    v8f h;
    #pragma unroll
    for (int k = 0; k < 8; ++k) {
        float i = sigm(gi[k]);
        float f = sigm(gf[k]);
        float g = tanh_fast(gg[k]);
        float o = sigm(go[k]);
        float cn = f * c[k] + i * g;
        c[k] = cn;
        h[k] = o * tanh_fast(cn);
    }
    store_h_f16(ha, lane, h);                  // feeds next step + decoder
}

__global__ __launch_bounds__(BLOCK_THREADS)
__attribute__((amdgpu_waves_per_eu(4)))       // <=256 VGPRs, 4 waves/SIMD
void lstm_ae_fused_kernel(
    const float* __restrict__ x,
    const float* __restrict__ Wih_r, const float* __restrict__ Whh_r,
    const float* __restrict__ bih_r, const float* __restrict__ bhh_r,
    const float* __restrict__ Wih_c, const float* __restrict__ Whh_c,
    const float* __restrict__ bih_c, const float* __restrict__ bhh_c,
    const float* __restrict__ Wih_d, const float* __restrict__ Whh_d,
    const float* __restrict__ bih_d, const float* __restrict__ bhh_d,
    const float* __restrict__ Wcls,  const float* __restrict__ bcls,
    float* __restrict__ out_dec, float* __restrict__ out_cls)
{
    extern __shared__ char smem[];
    _Float16* s_wih_r = (_Float16*)(smem + OFF_WIH_R);
    _Float16* s_whh_r = (_Float16*)(smem + OFF_WHH_R);
    _Float16* s_wih_c = (_Float16*)(smem + OFF_WIH_C);
    _Float16* s_whh_c = (_Float16*)(smem + OFF_WHH_C);
    _Float16* s_wih_d = (_Float16*)(smem + OFF_WIH_D);
    _Float16* s_whh_d = (_Float16*)(smem + OFF_WHH_D);
    float* s_bias_r = (float*)(smem + OFF_BIAS_R);
    float* s_bias_c = (float*)(smem + OFF_BIAS_C);
    float* s_bias_d = (float*)(smem + OFF_BIAS_D);

    const int tid = threadIdx.x;

    // ---- stage weights once: fp32 -> fp16, K padded to 32 with zeros ----
    for (int i = tid; i < 64 * KPAD; i += BLOCK_THREADS) {
        int n = i >> 5, k = i & 31;
        s_wih_r[i] = (_Float16)((k < IDIM) ? Wih_r[n * IDIM + k] : 0.0f);
        s_whh_r[i] = (_Float16)((k < HDIM) ? Whh_r[n * HDIM + k] : 0.0f);
        s_wih_c[i] = (_Float16)((k < IDIM) ? Wih_c[n * IDIM + k] : 0.0f);
        s_whh_c[i] = (_Float16)((k < HDIM) ? Whh_c[n * HDIM + k] : 0.0f);
    }
    for (int i = tid; i < 112 * KPAD; i += BLOCK_THREADS) {
        int n = i >> 5, k = i & 31;
        s_wih_d[i] = (_Float16)((k < HDIM) ? Wih_d[n * HDIM + k] : 0.0f);
        s_whh_d[i] = (_Float16)((k < IDIM) ? Whh_d[n * IDIM + k] : 0.0f);
    }
    for (int i = tid; i < 64; i += BLOCK_THREADS) {
        s_bias_r[i] = bih_r[i] + bhh_r[i];
        s_bias_c[i] = bih_c[i] + bhh_c[i];
    }
    for (int i = tid; i < 112; i += BLOCK_THREADS)
        s_bias_d[i] = bih_d[i] + bhh_d[i];

    // ---- per-wave staging ----
    const int w = tid >> 5, lane = tid & 31;
    char* stg = smem + OFF_STAGE + (unsigned)w * STAGE_PER_WAVE;
    _Float16* xa0  = (_Float16*)(stg + 0);      // x tile ping [16][32] f16
    _Float16* xa1  = (_Float16*)(stg + 1024);   // x tile pong [16][32] f16
    _Float16* ha_r = (_Float16*)(stg + 2048);   // enc_rec h  [16][32] f16
    _Float16* ha_c = (_Float16*)(stg + 3072);   // enc_clas h [16][32] f16
    _Float16* ha_d = (_Float16*)(stg + 4096);   // decoder h  [16][32] f16
    float*    gts  = (float*)(stg + 5120);      // decoder gates [16][112] f32
    float*    cd   = (float*)(stg + 12288);     // decoder cell  [16][28]  f32

    // zero this wave's staging (h0=c0=0 and K-padding columns)
    {
        unsigned int* p = (unsigned int*)stg;
        for (int i = lane; i < (int)(STAGE_PER_WAVE / 4); i += 32) p[i] = 0u;
    }
    __syncthreads();

    const int b0   = (blockIdx.x * NWAVES + w) * TILE_M;
    const int nloc = lane & 15;
    const int row0 = (lane >> 4) * 8;

    // ---- hoist loop-invariant bias scalars into registers (15 VGPRs) ----
    float br[4], bc[4], bd[7];
    #pragma unroll
    for (int g = 0; g < 4; ++g) {
        br[g] = s_bias_r[g * HDIM + nloc];
        bc[g] = s_bias_c[g * HDIM + nloc];
    }
    #pragma unroll
    for (int nt = 0; nt < 7; ++nt) bd[nt] = s_bias_d[nt * 16 + nloc];

    v8f c_r = splat8(0.0f);
    v8f c_c = splat8(0.0f);

    // ---- software-pipelined scan over T ----
    // Iteration t computes from the x tile staged during t-1 (LDS ping-pong);
    // global loads for t+1 are issued early and land in LDS at end of body.
    float xr[XREG];
    fetch_x(x, b0, 0, lane, xr);
    stage_x(xa0, lane, xr);

    for (int t = 0; t < TSTEPS; ++t) {
        _Float16* xa_cur = (t & 1) ? xa1 : xa0;
        _Float16* xa_nxt = (t & 1) ? xa0 : xa1;

        const bool more = (t + 1 < TSTEPS);
        if (more) fetch_x(x, b0, t + 1, lane, xr);   // overlaps full timestep

        v16h aX = load_A_frag(xa_cur, lane);
        enc_step(aX, ha_r, s_wih_r, s_whh_r, br, c_r, lane);   // enc_rec
        enc_step(aX, ha_c, s_wih_c, s_whh_c, bc, c_c, lane);   // enc_clas

        // ---- decoder: input = enc_rec h (this t), recurrent = dec h (t-1) ----
        v16h aE = load_A_frag(ha_r, lane);
        v16h aH = load_A_frag(ha_d, lane);
        #pragma unroll
        for (int nt = 0; nt < 7; ++nt) {             // 4*I = 112 = 7 N-tiles
            v8f acc = splat8(bd[nt]);
            acc = wmma_f16(aE, load_B_frag(s_wih_d, nt * 16, lane), acc);
            acc = wmma_f16(aH, load_B_frag(s_whh_d, nt * 16, lane), acc);
            #pragma unroll
            for (int k = 0; k < 8; ++k)
                gts[(row0 + k) * 112 + nt * 16 + nloc] = acc[k];
        }
        // gate tiles (16) misalign with gate width (28): elementwise via LDS.
        for (int idx = lane; idx < TILE_M * IDIM; idx += 32) {   // 448 = 14*32
            int r = idx / IDIM, n = idx - r * IDIM;
            const float* gr = gts + r * 112;
            float ig = sigm(gr[n]);
            float fg = sigm(gr[28 + n]);
            float gg = tanh_fast(gr[56 + n]);
            float og = sigm(gr[84 + n]);
            float cn = fg * cd[r * IDIM + n] + ig * gg;
            cd[r * IDIM + n] = cn;
            float h = og * tanh_fast(cn);
            ha_d[r * KPAD + n] = (_Float16)h;        // recurrent input for t+1
            out_dec[((size_t)(b0 + r) * TSTEPS + t) * IDIM + n] = h;
        }

        if (more) stage_x(xa_nxt, lane, xr);         // land prefetch for t+1
    }

    // ---- classifier on final enc_clas hidden (160 = 5*32: no divergence) ----
    for (int idx = lane; idx < TILE_M * NCLS; idx += 32) {
        int r = idx / NCLS, c = idx - r * NCLS;
        float acc = bcls[c];
        #pragma unroll
        for (int k = 0; k < HDIM; ++k)
            acc += (float)ha_c[r * KPAD + k] * Wcls[c * HDIM + k];
        out_cls[(size_t)(b0 + r) * NCLS + c] = acc;
    }
}

extern "C" void kernel_launch(void* const* d_in, const int* in_sizes, int n_in,
                              void* d_out, int out_size, void* d_ws, size_t ws_size,
                              hipStream_t stream) {
    (void)in_sizes; (void)n_in; (void)out_size; (void)d_ws; (void)ws_size;

    const float* x     = (const float*)d_in[0];
    const float* Wih_r = (const float*)d_in[1];
    const float* Whh_r = (const float*)d_in[2];
    const float* bih_r = (const float*)d_in[3];
    const float* bhh_r = (const float*)d_in[4];
    const float* Wih_c = (const float*)d_in[5];
    const float* Whh_c = (const float*)d_in[6];
    const float* bih_c = (const float*)d_in[7];
    const float* bhh_c = (const float*)d_in[8];
    const float* Wih_d = (const float*)d_in[9];
    const float* Whh_d = (const float*)d_in[10];
    const float* bih_d = (const float*)d_in[11];
    const float* bhh_d = (const float*)d_in[12];
    const float* Wcls  = (const float*)d_in[13];
    const float* bcls  = (const float*)d_in[14];

    float* out_dec = (float*)d_out;
    float* out_cls = out_dec + (size_t)BATCH * TSTEPS * IDIM;   // tuple: (dec, classification)

    (void)hipFuncSetAttribute((const void*)lstm_ae_fused_kernel,
                              hipFuncAttributeMaxDynamicSharedMemorySize,
                              (int)SMEM_BYTES);

    dim3 grid(BATCH / (TILE_M * NWAVES));   // 256 blocks
    dim3 block(BLOCK_THREADS);              // 8 waves/block, 1 batch tile/wave
    lstm_ae_fused_kernel<<<grid, block, SMEM_BYTES, stream>>>(
        x, Wih_r, Whh_r, bih_r, bhh_r,
        Wih_c, Whh_c, bih_c, bhh_c,
        Wih_d, Whh_d, bih_d, bhh_d,
        Wcls, bcls, out_dec, out_cls);
}
